// GDN_24936580120741
// MI455X (gfx1250) — matrix operand order repaired and checked by reading, not compiled, not run on hardware
//
#include <hip/hip_runtime.h>
#include <math.h>

// MI455X / gfx1250, wave32. Pipeline:
//  prep:   normalize embedding -> ne16 (f16), e_i = emb@att[0:64], e_j = emb@att[128:192]
//  cvt_w1: w1 -> f16
//  xh:     xh[n,b,:] = lin_w @ x[b,n,:]; xi = xh@att[64:128], xj = xh@att[192:256]
//  sim_topk: sim = ne@ne^T via v_wmma_f32_16x16x32_f16 into 256KB LDS slab, top-15/row
//  gcn:    leaky_relu attention logits, softmax over K, gather-weighted sum, *emb -> out16 (f16)
//  mlp:    w1h staged to LDS via TDM tensor_load_to_lds; fused
//          relu(out16 @ w1^T + b1) @ w2 + b2 via WMMA + wave reduction -> y (B,N)

typedef _Float16 h16;
typedef __attribute__((ext_vector_type(16))) _Float16 v16h;
typedef __attribute__((ext_vector_type(8)))  _Float16 v8h;
typedef __attribute__((ext_vector_type(8)))  float    v8f;
typedef __attribute__((ext_vector_type(4)))  unsigned int v4u;
typedef __attribute__((ext_vector_type(8)))  unsigned int v8u;

#define NN 4096
#define KK 15
#define DD 64
#define HH 256
#define BB 32

// ---- WMMA helpers -----------------------------------------------------------

__device__ inline v8f wmma_f16(v16h a, v16h b, v8f c) {
  return __builtin_amdgcn_wmma_f32_16x16x32_f16(false, a, false, b, (short)0, c,
                                                false, false);
}

// A tile 16x32 (MxK), row-major source, row_stride in halves.
// lane<16: row=lane, K 0..15; lane>=16: same row, K 16..31.
__device__ inline v16h load_a_tile(const h16* base, int row_stride, int lane, int kblock) {
  int m = lane & 15;
  int c = ((lane >> 4) << 4) + kblock;
  const h16* p = base + m * row_stride + c;
  return *(const v16h*)(p);
}

// B tile 32x16 (KxN) where B[k][n] = W[col_n][k], W row-major (row_stride halves).
// lane holds column lane&15, K = (lane>>4)*16 + 0..15 (contiguous 32B load).
__device__ inline v16h load_b_tile(const h16* base, int row_stride, int lane, int kblock) {
  int n = lane & 15;
  int c = ((lane >> 4) << 4) + kblock;
  return *(const v16h*)(base + n * row_stride + c);
}

// ---- Kernel A: normalize embedding + e_i/e_j dots ---------------------------

__global__ __launch_bounds__(256)
void prep_kernel(const float* __restrict__ emb, const float* __restrict__ att,
                 h16* __restrict__ ne16, float* __restrict__ e_i, float* __restrict__ e_j) {
  int wave = threadIdx.x >> 5, lane = threadIdx.x & 31;
  int row = blockIdx.x * 8 + wave;
  float2 e = *(const float2*)(emb + row * DD + lane * 2);
  float ss = e.x * e.x + e.y * e.y;
  float di = e.x * att[2 * lane] + e.y * att[2 * lane + 1];
  float dj = e.x * att[128 + 2 * lane] + e.y * att[128 + 2 * lane + 1];
#pragma unroll
  for (int off = 16; off >= 1; off >>= 1) {
    ss += __shfl_xor(ss, off, 32);
    di += __shfl_xor(di, off, 32);
    dj += __shfl_xor(dj, off, 32);
  }
  float inv = 1.0f / (sqrtf(ss) + 1e-8f);
  ne16[row * DD + 2 * lane]     = (h16)(e.x * inv);
  ne16[row * DD + 2 * lane + 1] = (h16)(e.y * inv);
  if (lane == 0) { e_i[row] = di; e_j[row] = dj; }
}

__global__ void cvt_w1_kernel(const float* __restrict__ w1, h16* __restrict__ w1h) {
  int i = blockIdx.x * 256 + threadIdx.x;
  w1h[i] = (h16)w1[i];
}

// ---- Kernel B: xh = einsum('bni,oi->nbo') + xi/xj dots ----------------------

__global__ __launch_bounds__(256)
void xh_kernel(const float* __restrict__ x, const float* __restrict__ lin_w,
               const float* __restrict__ att, float* __restrict__ xh,
               float* __restrict__ xi, float* __restrict__ xj) {
  __shared__ float lw[DD * 5];
  __shared__ float axi[DD], axj[DD];
  int t = threadIdx.x;
  if (t < DD * 5) lw[t] = lin_w[t];
  if (t < DD) { axi[t] = att[64 + t]; axj[t] = att[192 + t]; }
  __syncthreads();
  int p = blockIdx.x * 256 + t;
  int n = p >> 5, b = p & 31;
  float xv[5];
#pragma unroll
  for (int i = 0; i < 5; ++i) xv[i] = x[(b * NN + n) * 5 + i];
  float si = 0.f, sj = 0.f;
  float* dst = xh + (n * BB + b) * DD;
#pragma unroll
  for (int o = 0; o < DD; ++o) {
    float acc = 0.f;
#pragma unroll
    for (int i = 0; i < 5; ++i) acc += xv[i] * lw[o * 5 + i];
    dst[o] = acc;
    si += acc * axi[o];
    sj += acc * axj[o];
  }
  xi[n * BB + b] = si;
  xj[n * BB + b] = sj;
}

// ---- Kernel C: sim = ne@ne^T (WMMA) into LDS, top-15 per row ----------------

__global__ __launch_bounds__(256)
void sim_topk_kernel(const h16* __restrict__ ne16, int* __restrict__ src) {
  extern __shared__ float simS[];  // 16 * 4096 floats = 256 KB
  int t = threadIdx.x, wave = t >> 5, lane = t & 31;
  int rowbase = blockIdx.x * 16;

  v16h a0 = load_a_tile(ne16 + rowbase * DD, DD, lane, 0);
  v16h a1 = load_a_tile(ne16 + rowbase * DD, DD, lane, 32);

  for (int ct = wave; ct < NN / 16; ct += 8) {
    v16h b0 = load_b_tile(ne16 + ct * 16 * DD, DD, lane, 0);
    v16h b1 = load_b_tile(ne16 + ct * 16 * DD, DD, lane, 32);
    v8f c = {};
    c = wmma_f16(a0, b0, c);
    c = wmma_f16(a1, b1, c);
    int m0 = (lane >> 4) * 8;
    int col = ct * 16 + (lane & 15);
#pragma unroll
    for (int r = 0; r < 8; ++r) simS[(m0 + r) * NN + col] = c[r];
  }
  __syncthreads();

  // wave w extracts top-15 for rows 2w and 2w+1 (iterative argmax; ties -> lower index)
#pragma unroll 1
  for (int rr = 0; rr < 2; ++rr) {
    int row = wave * 2 + rr;
    volatile float* sr = simS + row * NN;
#pragma unroll 1
    for (int k = 0; k < KK; ++k) {
      float best = -INFINITY;
      int bidx = 0x7fffffff;
#pragma unroll 4
      for (int j = lane; j < NN; j += 32) {
        float v = sr[j];
        if (v > best) { best = v; bidx = j; }
      }
#pragma unroll
      for (int off = 16; off >= 1; off >>= 1) {
        float ov = __shfl_xor(best, off, 32);
        int   oi = __shfl_xor(bidx, off, 32);
        if (ov > best || (ov == best && oi < bidx)) { best = ov; bidx = oi; }
      }
      if (lane == (bidx & 31)) sr[bidx] = -INFINITY;
      if (lane == 0) src[(rowbase + row) * KK + k] = bidx;
    }
  }
}

// ---- Kernel D: attention logits, softmax over K, gather-weighted sum --------

__global__ __launch_bounds__(512)
void gcn_kernel(const float* __restrict__ xh, const int* __restrict__ src,
                const float* __restrict__ e_i, const float* __restrict__ e_j,
                const float* __restrict__ xi, const float* __restrict__ xj,
                const float* __restrict__ emb, h16* __restrict__ out16) {
  __shared__ int   sK[KK];
  __shared__ float ejK[KK];
  __shared__ float alpha[KK][BB + 1];
  int n = blockIdx.x, t = threadIdx.x;
  if (t < KK) {
    int s = src[n * KK + t];
    sK[t] = s;
    ejK[t] = e_j[s];
    __builtin_prefetch(xh + (size_t)s * BB * DD, 0, 0);  // global_prefetch_b8
  }
  __syncthreads();
  if (t < KK * BB) {
    int k = t >> 5, b = t & 31;
    float v = e_i[n] + xi[n * BB + b] + ejK[k] + xj[sK[k] * BB + b];
    v = v > 0.f ? v : 0.2f * v;  // leaky_relu(0.2)
    alpha[k][b] = v;
  }
  __syncthreads();
  if (t < BB) {
    int b = t;
    float m = -INFINITY;
#pragma unroll
    for (int k = 0; k < KK; ++k) m = fmaxf(m, alpha[k][b]);
    float ex[KK], s = 0.f;
#pragma unroll
    for (int k = 0; k < KK; ++k) { ex[k] = __expf(alpha[k][b] - m); s += ex[k]; }
    float inv = 1.f / s;
#pragma unroll
    for (int k = 0; k < KK; ++k) alpha[k][b] = ex[k] * inv;
  }
  __syncthreads();
#pragma unroll
  for (int i = 0; i < 4; ++i) {
    int p = t + i * 512;       // p over (b,d): B*D = 2048
    int b = p >> 6, d = p & 63;
    float acc = 0.f;
#pragma unroll
    for (int k = 0; k < KK; ++k)
      acc += alpha[k][b] * xh[(sK[k] * BB + b) * DD + d];
    acc *= emb[n * DD + d];
    out16[((b << 12) + n) * DD + d] = (h16)acc;  // row m = b*N + n
  }
}

// ---- Kernel E: fused relu(out@w1^T + b1) @ w2 + b2 (WMMA) -------------------
// w1h (32 KB) is staged into LDS once per block with the Tensor Data Mover.

__global__ __launch_bounds__(256)
void mlp_kernel(const h16* __restrict__ out16, const h16* __restrict__ w1h,
                const float* __restrict__ b1, const float* __restrict__ w2,
                const float* __restrict__ b2, float* __restrict__ y) {
  __shared__ __align__(32) h16 sw1[HH * DD];  // 32 KB, LDS offset 0
  int t = threadIdx.x, wave = t >> 5, lane = t & 31;
  int rowbase = (blockIdx.x * 8 + wave) * 16;

  if (wave == 0) {
    // Tensor DMA descriptor (ISA ch.8): 1 tile of 16384 x 2-byte elements,
    // global w1h -> LDS offset 0. Group2/3 omitted (<=2D tensor).
    unsigned long long ga = (unsigned long long)(const void*)w1h;
    v4u g0;
    g0.x = 1u;                                        // count=1 (valid), user mode
    g0.y = 0u;                                        // lds_addr = 0 (sw1)
    g0.z = (unsigned)(ga & 0xffffffffull);            // global_addr[31:0]
    g0.w = (unsigned)((ga >> 32) & 0x1ffffffull) | (2u << 30);  // addr[56:32], type=2
    v8u g1;
    g1[0] = (1u << 16);                 // workgroup_mask=0, data_size=1 (2 bytes)
    g1[1] = ((HH * DD) & 0xffffu) << 16;          // tensor_dim0[15:0]
    g1[2] = (((HH * DD) >> 16) & 0xffffu) | (1u << 16); // tensor_dim0[31:16], tensor_dim1=1
    g1[3] = ((unsigned)(HH * DD)) << 16;          // tile_dim0 = 16384
    g1[4] = 1u;                                   // tile_dim1 = 1, tile_dim2 unused
    g1[5] = (unsigned)(HH * DD);                  // tensor_dim0_stride[31:0]
    g1[6] = 0u;
    g1[7] = 0u;
    asm volatile("tensor_load_to_lds %0, %1" :: "s"(g0), "s"(g1) : "memory");
    __builtin_amdgcn_s_wait_tensorcnt(0);
  }
  __syncthreads();

  v16h a0 = load_a_tile(out16 + rowbase * DD, DD, lane, 0);
  v16h a1 = load_a_tile(out16 + rowbase * DD, DD, lane, 32);

  float acc[8];
#pragma unroll
  for (int r = 0; r < 8; ++r) acc[r] = 0.f;

  for (int ht = 0; ht < HH / 16; ++ht) {
    v16h b0t = load_b_tile(sw1 + ht * 16 * DD, DD, lane, 0);   // ds loads
    v16h b1t = load_b_tile(sw1 + ht * 16 * DD, DD, lane, 32);
    v8f c = {};
    c = wmma_f16(a0, b0t, c);
    c = wmma_f16(a1, b1t, c);
    int h = ht * 16 + (lane & 15);
    float bias = b1[h], w = w2[h];
#pragma unroll
    for (int r = 0; r < 8; ++r) {
      float hv = c[r] + bias;
      hv = hv > 0.f ? hv : 0.f;   // relu
      acc[r] += hv * w;           // h @ w2
    }
  }
  // reduce the 16 H-columns held across lanes within each 16-lane half
#pragma unroll
  for (int off = 8; off >= 1; off >>= 1)
#pragma unroll
    for (int r = 0; r < 8; ++r) acc[r] += __shfl_xor(acc[r], off, 32);

  if ((lane & 15) == 0) {
    int mbase = rowbase + (lane >> 4) * 8;  // C layout: row = r + 8*(lane>>4)
    float bb = b2[0];
#pragma unroll
    for (int r = 0; r < 8; ++r) y[mbase + r] = acc[r] + bb;
  }
}

// ---- launch -----------------------------------------------------------------

extern "C" void kernel_launch(void* const* d_in, const int* in_sizes, int n_in,
                              void* d_out, int out_size, void* d_ws, size_t ws_size,
                              hipStream_t stream) {
  const float* x     = (const float*)d_in[0];
  const float* emb   = (const float*)d_in[1];
  const float* lin_w = (const float*)d_in[2];
  const float* att   = (const float*)d_in[3];
  const float* w1    = (const float*)d_in[4];
  const float* b1    = (const float*)d_in[5];
  const float* w2    = (const float*)d_in[6];
  const float* b2    = (const float*)d_in[7];
  float* y = (float*)d_out;

  char* ws = (char*)d_ws;   // ~52 MB total
  h16*   ne16  = (h16*)(ws + 0);          // 4096*64*2   = 512 KB
  h16*   w1h   = (h16*)(ws + 524288);     // 256*64*2    = 32 KB
  float* e_i   = (float*)(ws + 557056);   // 16 KB
  float* e_j   = (float*)(ws + 573440);   // 16 KB
  float* xi    = (float*)(ws + 589824);   // 4096*32*4   = 512 KB
  float* xj    = (float*)(ws + 1114112);  // 512 KB
  int*   src   = (int*)(ws + 1638400);    // 4096*15*4   = 240 KB
  float* xh    = (float*)(ws + 1884160);  // 4096*32*64*4 = 32 MB
  h16*   out16 = (h16*)(ws + 35438592);   // 131072*64*2 = 16 MB

  prep_kernel<<<NN / 8, 256, 0, stream>>>(emb, att, ne16, e_i, e_j);
  cvt_w1_kernel<<<(HH * DD) / 256, 256, 0, stream>>>(w1, w1h);
  xh_kernel<<<(NN * BB) / 256, 256, 0, stream>>>(x, lin_w, att, xh, xi, xj);
  sim_topk_kernel<<<NN / 16, 256, 16 * NN * 4, stream>>>(ne16, src);
  gcn_kernel<<<NN, 512, 0, stream>>>(xh, src, e_i, e_j, xi, xj, emb, out16);
  mlp_kernel<<<(NN * BB) / 128, 256, 0, stream>>>(out16, w1h, b1, w2, b2, y);
}
